// SlotAttention_38706245271647
// MI455X (gfx1250) — matrix operand order repaired and checked by reading, not compiled, not run on hardware
//
#include <hip/hip_runtime.h>
#include <hip/hip_bf16.h>

// ---------------------------------------------------------------------------
// Slot Attention for MI455X (gfx1250, wave32).
// Heavy GEMM (k/v projection, 34.4 GFLOP) runs on v_wmma_f32_16x16x32_bf16
// with f32 accumulate; everything feeding the f32 outputs stays f32.
// ---------------------------------------------------------------------------

typedef __bf16 bf16_t;
typedef __attribute__((ext_vector_type(16))) __bf16 v16bf;
typedef __attribute__((ext_vector_type(8)))  __bf16 v8bf;
typedef __attribute__((ext_vector_type(8)))  float  v8f;

constexpr int B = 32, T = 4096, C = 256, S = 8, D = 256, H = 512;
constexpr int NKV = 2 * D;                 // k and v projections fused: 512 output cols
constexpr float LN_EPS_F = 1e-5f;
constexpr float EPS_F    = 1e-8f;

// ---------------- LayerNorm over rows of length 256 ----------------
template <typename OUT>
__global__ __launch_bounds__(256) void ln_rows(const float* __restrict__ in,
                                               const float* __restrict__ g,
                                               const float* __restrict__ bb,
                                               OUT* __restrict__ out) {
  __shared__ float red[256];
  const int row = blockIdx.x;
  const int c = threadIdx.x;
  float x = in[(size_t)row * 256 + c];
  red[c] = x;
  __syncthreads();
  for (int k = 128; k > 0; k >>= 1) { if (c < k) red[c] += red[c + k]; __syncthreads(); }
  float m = red[0] * (1.0f / 256.0f);
  __syncthreads();
  float d = x - m;
  red[c] = d * d;
  __syncthreads();
  for (int k = 128; k > 0; k >>= 1) { if (c < k) red[c] += red[c + k]; __syncthreads(); }
  float inv = rsqrtf(red[0] * (1.0f / 256.0f) + LN_EPS_F);
  out[(size_t)row * 256 + c] = (OUT)(d * inv * g[c] + bb[c]);
}

// ---------------- pack Wk|Wv into one bf16 weight matrix [512,256] ----------
__global__ __launch_bounds__(256) void pack_w_bf16(const float* __restrict__ Wk,
                                                   const float* __restrict__ Wv,
                                                   bf16_t* __restrict__ W) {
  int idx = blockIdx.x * 256 + threadIdx.x;   // 512*256
  float f = (idx < D * C) ? Wk[idx] : Wv[idx - D * C];
  W[idx] = (bf16_t)f;
}

// ---------------- k/v projection: [B*T,256] @ [256,512]^T via WMMA ----------
// Block = 256 thr = 8 waves; block owns a 16-row M tile; wave w owns N range
// [w*64, w*64+64) as 4 accumulators. A/B use contiguous k = 16*half + s; C/D per ISA 7.12.2.
__global__ __launch_bounds__(256) void kv_gemm_wmma(const bf16_t* __restrict__ X,
                                                    const bf16_t* __restrict__ W,
                                                    float* __restrict__ Kout,
                                                    float* __restrict__ Vout) {
  const int lane = threadIdx.x & 31;
  const int wave = threadIdx.x >> 5;          // 0..7
  const int half = lane >> 4;                 // 0/1 -> lane group
  const int l15  = lane & 15;
  const size_t mbase = (size_t)blockIdx.x * 16;

  v8f acc[4];
#pragma unroll
  for (int j = 0; j < 4; ++j) acc[j] = (v8f){0.f,0.f,0.f,0.f,0.f,0.f,0.f,0.f};

  // A row pointer for this lane: M = mbase + l15, lane-half selects K+0..15/16..31
  const bf16_t* arow = X + (mbase + l15) * C + half * 16;

  for (int kb = 0; kb < C; kb += 32) {
    // A fragment: elems 0..15 = K kb+half*16+0..15
    v8bf alo = *(const v8bf*)(arow + kb);
    v8bf ahi = *(const v8bf*)(arow + kb + 8);
    v16bf a;
#pragma unroll
    for (int i = 0; i < 8; ++i) { a[i] = alo[i]; a[8 + i] = ahi[i]; }

#pragma unroll
    for (int j = 0; j < 4; ++j) {
      // B fragment: N = lane&15, lane-half selects K group; 16 contiguous K per lane
      int n = wave * 64 + j * 16 + l15;
      v16bf bfrag = *(const v16bf*)(W + (size_t)n * C + kb + half * 16);
      acc[j] = __builtin_amdgcn_wmma_f32_16x16x32_bf16(
          false, a, false, bfrag, (short)0, acc[j], false, false);
    }
  }

  // C/D layout: VGPR r -> M = mbase + r + 8*half, N = l15 (lanes 0-15 / 16-31)
#pragma unroll
  for (int j = 0; j < 4; ++j) {
    int n = wave * 64 + j * 16 + l15;
    float* dst = (n < D) ? Kout : Vout;
    int nn = n & (D - 1);
#pragma unroll
    for (int r = 0; r < 8; ++r) {
      size_t m = mbase + (size_t)r + 8 * half;
      dst[m * D + nn] = acc[j][r];
    }
  }
}

// ---------------- q = LN(slots) @ Wq^T * D^-0.5 ----------------
__global__ __launch_bounds__(256) void qproj(const float* __restrict__ sn,
                                             const float* __restrict__ Wq,
                                             float* __restrict__ q) {
  int idx = blockIdx.x * 256 + threadIdx.x;   // B*S*D
  int r = idx >> 8, j = idx & 255;
  const float* x = sn + (size_t)r * D;
  const float* w = Wq + (size_t)j * D;
  float a = 0.f;
#pragma unroll 4
  for (int k = 0; k < D; ++k) a += x[k] * w[k];
  q[idx] = a * 0.0625f;                       // 256^-0.5
}

// ---------------- logits[b,s,t] = q[b,s,:] . k[b,t,:] (wave per token) ------
__global__ __launch_bounds__(256) void logits_kernel(const float* __restrict__ q,
                                                     const float* __restrict__ k,
                                                     float* __restrict__ logits) {
  const int lane = threadIdx.x & 31;
  const int wave = threadIdx.x >> 5;
  const int t = blockIdx.x * 8 + wave;
  const int b = blockIdx.y;
  const float* krow = k + ((size_t)b * T + t) * D + lane * 8;
  float kk[8];
#pragma unroll
  for (int i = 0; i < 8; ++i) kk[i] = krow[i];
  const float* qb = q + (size_t)b * S * D + lane * 8;
#pragma unroll
  for (int s = 0; s < S; ++s) {
    float p = 0.f;
#pragma unroll
    for (int i = 0; i < 8; ++i) p += qb[s * D + i] * kk[i];
    for (int off = 16; off > 0; off >>= 1) p += __shfl_down(p, off, 32);
    if (lane == 0) logits[((size_t)b * S + s) * T + t] = p;
  }
}

// ---------------- softmax over S=8 slots, +EPS (pre L1-renorm) --------------
__global__ __launch_bounds__(256) void softmax_slots(const float* __restrict__ logits,
                                                     float* __restrict__ attn) {
  int idx = blockIdx.x * 256 + threadIdx.x;   // B*T
  int b = idx / T, t = idx % T;
  const float* base = logits + (size_t)b * S * T + t;
  float vals[S];
  float mx = -3.4e38f;
#pragma unroll
  for (int s = 0; s < S; ++s) { vals[s] = base[(size_t)s * T]; mx = fmaxf(mx, vals[s]); }
  float sum = 0.f;
#pragma unroll
  for (int s = 0; s < S; ++s) { vals[s] = expf(vals[s] - mx); sum += vals[s]; }
  float inv = 1.0f / sum;
  float* ob = attn + (size_t)b * S * T + t;
#pragma unroll
  for (int s = 0; s < S; ++s) ob[(size_t)s * T] = vals[s] * inv + EPS_F;
}

// ---------------- per-(b,s) token sums --------------------------------------
__global__ __launch_bounds__(256) void rowsum(const float* __restrict__ attn,
                                              float* __restrict__ sums) {
  __shared__ float red[256];
  const int bs = blockIdx.x, tid = threadIdx.x;
  const float* row = attn + (size_t)bs * T;
  float s = 0.f;
  for (int t = tid; t < T; t += 256) s += row[t];
  red[tid] = s;
  __syncthreads();
  for (int k = 128; k > 0; k >>= 1) { if (tid < k) red[tid] += red[tid + k]; __syncthreads(); }
  if (tid == 0) sums[bs] = red[0];
}

__global__ __launch_bounds__(256) void renorm(float* __restrict__ attn,
                                              const float* __restrict__ sums) {
  size_t idx = (size_t)blockIdx.x * 256 + threadIdx.x;  // B*S*T
  attn[idx] = attn[idx] / sums[idx / T];
}

// ---------------- updates[b,s,d] = sum_t attn[b,s,t]*v[b,t,d] ---------------
__global__ __launch_bounds__(256) void updates_kernel(const float* __restrict__ attn,
                                                      const float* __restrict__ v,
                                                      float* __restrict__ upd) {
  __shared__ float sm[4 * S * 64];
  const int b = blockIdx.x >> 2;
  const int dbase = (blockIdx.x & 3) * 64;
  const int dloc = threadIdx.x & 63;
  const int tq = threadIdx.x >> 6;            // 0..3 t-stripes
  float acc[S] = {};
  for (int t = tq; t < T; t += 4) {
    if (t + 64 < T)  // gfx1250 global_prefetch_b8: stream next v lines
      __builtin_prefetch(&v[((size_t)b * T + t + 64) * D + dbase + dloc], 0, 0);
    float vv = v[((size_t)b * T + t) * D + dbase + dloc];
#pragma unroll
    for (int s = 0; s < S; ++s) acc[s] += attn[((size_t)b * S + s) * T + t] * vv;
  }
#pragma unroll
  for (int s = 0; s < S; ++s) sm[(tq * S + s) * 64 + dloc] = acc[s];
  __syncthreads();
  if (tq == 0) {
#pragma unroll
    for (int s = 0; s < S; ++s) {
      float tot = sm[(0 * S + s) * 64 + dloc] + sm[(1 * S + s) * 64 + dloc] +
                  sm[(2 * S + s) * 64 + dloc] + sm[(3 * S + s) * 64 + dloc];
      upd[((size_t)b * S + s) * D + dbase + dloc] = tot;
    }
  }
}

// ---------------- GRU gate projections --------------------------------------
__global__ __launch_bounds__(256) void gru_gates(const float* __restrict__ upd,
                                                 const float* __restrict__ h,
                                                 const float* __restrict__ Wih,
                                                 const float* __restrict__ Whh,
                                                 const float* __restrict__ bih,
                                                 const float* __restrict__ bhh,
                                                 float* __restrict__ gx,
                                                 float* __restrict__ gh) {
  int idx = blockIdx.x * 256 + threadIdx.x;   // B*S*3D
  int r = idx / (3 * D), c = idx % (3 * D);
  const float* xr = upd + (size_t)r * D;
  const float* hr = h + (size_t)r * D;
  const float* wi = Wih + (size_t)c * D;
  const float* wh = Whh + (size_t)c * D;
  float ax = bih[c], ah = bhh[c];
#pragma unroll 4
  for (int k = 0; k < D; ++k) { ax += xr[k] * wi[k]; ah += hr[k] * wh[k]; }
  gx[idx] = ax;
  gh[idx] = ah;
}

__global__ __launch_bounds__(256) void gru_combine(const float* __restrict__ gx,
                                                   const float* __restrict__ gh,
                                                   float* __restrict__ hio) {
  int idx = blockIdx.x * 256 + threadIdx.x;   // B*S*D
  int r = idx >> 8, d = idx & 255;
  const float* gxr = gx + (size_t)r * 3 * D;
  const float* ghr = gh + (size_t)r * 3 * D;
  float xr = gxr[d], xz = gxr[D + d], xn = gxr[2 * D + d];
  float hr = ghr[d], hz = ghr[D + d], hn = ghr[2 * D + d];
  float rg = 1.0f / (1.0f + expf(-(xr + hr)));
  float z  = 1.0f / (1.0f + expf(-(xz + hz)));
  float n  = tanhf(xn + rg * hn);
  float hp = hio[idx];
  hio[idx] = (1.0f - z) * n + z * hp;
}

// ---------------- MLP residual ---------------------------------------------
__global__ __launch_bounds__(256) void mlp1(const float* __restrict__ mn,
                                            const float* __restrict__ W1,
                                            const float* __restrict__ b1,
                                            float* __restrict__ h1) {
  int idx = blockIdx.x * 256 + threadIdx.x;   // B*S*H
  int r = idx / H, c = idx % H;
  const float* x = mn + (size_t)r * D;
  const float* w = W1 + (size_t)c * D;
  float a = b1[c];
#pragma unroll 4
  for (int k = 0; k < D; ++k) a += x[k] * w[k];
  h1[idx] = fmaxf(a, 0.0f);
}

__global__ __launch_bounds__(256) void mlp2(const float* __restrict__ h1,
                                            const float* __restrict__ W2,
                                            const float* __restrict__ b2,
                                            float* __restrict__ slots) {
  int idx = blockIdx.x * 256 + threadIdx.x;   // B*S*D
  int r = idx >> 8, d = idx & 255;
  const float* x = h1 + (size_t)r * H;
  const float* w = W2 + (size_t)d * H;
  float a = b2[d];
#pragma unroll 4
  for (int k = 0; k < H; ++k) a += x[k] * w[k];
  slots[idx] += a;
}

// ---------------------------------------------------------------------------
extern "C" void kernel_launch(void* const* d_in, const int* in_sizes, int n_in,
                              void* d_out, int out_size, void* d_ws, size_t ws_size,
                              hipStream_t stream) {
  (void)in_sizes; (void)n_in; (void)out_size; (void)ws_size;
  const float* inputs  = (const float*)d_in[0];
  const float* slots0  = (const float*)d_in[1];
  const float* ln_in_g = (const float*)d_in[2];
  const float* ln_in_b = (const float*)d_in[3];
  const float* ln_s_g  = (const float*)d_in[4];
  const float* ln_s_b  = (const float*)d_in[5];
  const float* ln_m_g  = (const float*)d_in[6];
  const float* ln_m_b  = (const float*)d_in[7];
  const float* Wq   = (const float*)d_in[8];
  const float* Wk   = (const float*)d_in[9];
  const float* Wv   = (const float*)d_in[10];
  const float* W_ih = (const float*)d_in[11];
  const float* W_hh = (const float*)d_in[12];
  const float* b_ih = (const float*)d_in[13];
  const float* b_hh = (const float*)d_in[14];
  const float* W1   = (const float*)d_in[15];
  const float* b1   = (const float*)d_in[16];
  const float* W2   = (const float*)d_in[17];
  const float* b2   = (const float*)d_in[18];

  float* out        = (float*)d_out;
  float* slots_out  = out;                         // [B,S,D]
  float* logits_out = out + (size_t)B * S * D;     // [B,S,T]
  float* attn_out   = logits_out + (size_t)B * S * T;  // [B,S,T]

  // workspace carve-out (~340 MB total)
  char* w = (char*)d_ws;
  size_t off = 0;
  auto take = [&](size_t bytes) -> char* {
    char* p = w + off;
    off = (off + bytes + 255) & ~(size_t)255;
    return p;
  };
  bf16_t* x_bf   = (bf16_t*)take((size_t)B * T * C * sizeof(bf16_t));
  bf16_t* wkv    = (bf16_t*)take((size_t)NKV * C * sizeof(bf16_t));
  float* kbuf    = (float*)take((size_t)B * T * D * sizeof(float));
  float* vbuf    = (float*)take((size_t)B * T * D * sizeof(float));
  float* s_norm  = (float*)take((size_t)B * S * D * sizeof(float));
  float* qbuf    = (float*)take((size_t)B * S * D * sizeof(float));
  float* sums    = (float*)take((size_t)B * S * sizeof(float));
  float* upd     = (float*)take((size_t)B * S * D * sizeof(float));
  float* gx      = (float*)take((size_t)B * S * 3 * D * sizeof(float));
  float* gh      = (float*)take((size_t)B * S * 3 * D * sizeof(float));
  float* m_norm  = (float*)take((size_t)B * S * D * sizeof(float));
  float* h1      = (float*)take((size_t)B * S * H * sizeof(float));
  float* slots_c = (float*)take((size_t)B * S * D * sizeof(float));

  // --- one-time preprocessing ---
  pack_w_bf16<<<(NKV * C) / 256, 256, 0, stream>>>(Wk, Wv, wkv);
  ln_rows<bf16_t><<<B * T, 256, 0, stream>>>(inputs, ln_in_g, ln_in_b, x_bf);
  kv_gemm_wmma<<<(B * T) / 16, 256, 0, stream>>>(x_bf, wkv, kbuf, vbuf);
  hipMemcpyAsync(slots_c, slots0, (size_t)B * S * D * sizeof(float),
                 hipMemcpyDeviceToDevice, stream);

  // --- 3 slot-attention iterations ---
  for (int it = 0; it < 3; ++it) {
    ln_rows<float><<<B * S, 256, 0, stream>>>(slots_c, ln_s_g, ln_s_b, s_norm);
    qproj<<<(B * S * D) / 256, 256, 0, stream>>>(s_norm, Wq, qbuf);
    dim3 lgrid(T / 8, B);
    logits_kernel<<<lgrid, 256, 0, stream>>>(qbuf, kbuf, logits_out);
    softmax_slots<<<(B * T) / 256, 256, 0, stream>>>(logits_out, attn_out);
    rowsum<<<B * S, 256, 0, stream>>>(attn_out, sums);
    renorm<<<(B * S * T) / 256, 256, 0, stream>>>(attn_out, sums);
    updates_kernel<<<B * 4, 256, 0, stream>>>(attn_out, vbuf, upd);
    gru_gates<<<(B * S * 3 * D) / 256, 256, 0, stream>>>(upd, slots_c, W_ih, W_hh,
                                                         b_ih, b_hh, gx, gh);
    gru_combine<<<(B * S * D) / 256, 256, 0, stream>>>(gx, gh, slots_c);
    ln_rows<float><<<B * S, 256, 0, stream>>>(slots_c, ln_m_g, ln_m_b, m_norm);
    mlp1<<<(B * S * H) / 256, 256, 0, stream>>>(m_norm, W1, b1, h1);
    mlp2<<<(B * S * D) / 256, 256, 0, stream>>>(h1, W2, b2, slots_c);
  }

  hipMemcpyAsync(slots_out, slots_c, (size_t)B * S * D * sizeof(float),
                 hipMemcpyDeviceToDevice, stream);
}
